// SlotAttentionAEv5_50036368999077
// MI455X (gfx1250) — compile-verified, matrix operand order checked by reading, not run on hardware
//
#include <hip/hip_runtime.h>
#include <hip/hip_bf16.h>

typedef __attribute__((ext_vector_type(16))) _Float16 v16h;
typedef __attribute__((ext_vector_type(8)))  _Float16 v8h;
typedef __attribute__((ext_vector_type(8)))  float    v8f;

#define BN   32
#define NN   4096
#define DD   64
#define KSL  10
#define MTOT (BN * NN)          // 131072 tokens

union AF { v16h v; v8h h[2]; };

__device__ inline v8h hzero8() {
  v8h z;
#pragma unroll
  for (int i = 0; i < 8; ++i) z[i] = (_Float16)0.0f;
  return z;
}

// ---------------------------------------------------------------------------
// conv1: direct fp32 conv (Cin=3), relu, write f16 channels-minor [m][64]
// ---------------------------------------------------------------------------
__global__ void conv1_kernel(const float* __restrict__ img,
                             const float* __restrict__ w,
                             const float* __restrict__ bias,
                             _Float16* __restrict__ out) {
  int idx = blockIdx.x * blockDim.x + threadIdx.x;   // MTOT*64
  if (idx >= MTOT * 64) return;
  int oc = idx & 63;
  int m  = idx >> 6;
  int b  = m >> 12;
  int n  = m & 4095;
  int y  = n >> 6, x = n & 63;
  float acc = bias[oc];
  for (int ci = 0; ci < 3; ++ci)
    for (int dy = 0; dy < 5; ++dy) {
      int sy = y + dy - 2;
      if ((unsigned)sy > 63u) continue;
      for (int dx = 0; dx < 5; ++dx) {
        int sx = x + dx - 2;
        if ((unsigned)sx > 63u) continue;
        acc = fmaf(img[(b * 3 + ci) * 4096 + sy * 64 + sx],
                   w[((oc * 3 + ci) * 5 + dy) * 5 + dx], acc);
      }
    }
  out[idx] = (_Float16)fmaxf(acc, 0.0f);
}

// convert conv weight (64,64,5,5) fp32 -> f16 layout [25][out64][in64]
__global__ void cvt_conv_w_kernel(const float* __restrict__ w, _Float16* __restrict__ wt) {
  int e = blockIdx.x * blockDim.x + threadIdx.x;   // 64*64*25
  if (e >= 64 * 64 * 25) return;
  int i = e & 63, o = (e >> 6) & 63, p = e >> 12;  // p = dy*5+dx
  wt[(p * 64 + o) * 64 + i] = (_Float16)w[(o * 64 + i) * 25 + p];
}

__global__ void cvt_f16_kernel(const float* __restrict__ src, _Float16* __restrict__ dst, int n) {
  int i = blockIdx.x * blockDim.x + threadIdx.x;
  if (i < n) dst[i] = (_Float16)src[i];
}

// ---------------------------------------------------------------------------
// conv2/3/4 as shifted-GEMM with WMMA. One wave = 16(M) x 64(N) tile.
// A: f16 activations [m][64]; Wt: f16 [25][o][i]; relu epilogue.
// ---------------------------------------------------------------------------
__global__ __launch_bounds__(128) void conv_wmma_kernel(
    const _Float16* __restrict__ A, const _Float16* __restrict__ Wt,
    const float* __restrict__ bias, _Float16* __restrict__ outH,
    float* __restrict__ outF) {
  int wave = threadIdx.x >> 5;
  int lane = threadIdx.x & 31;
  int tile = blockIdx.x * 4 + wave;
  int m0 = tile << 4;
  int b  = m0 >> 12;
  int y  = (m0 >> 6) & 63;
  int x0 = m0 & 63;
  int row = lane & 15;
  int kb  = (lane >> 4) << 3;

  __builtin_prefetch(Wt + lane * 128, 0, 1);   // warm weight panel (global_prefetch_b8)

  v8f acc[4];
#pragma unroll
  for (int t = 0; t < 4; ++t)
#pragma unroll
    for (int i = 0; i < 8; ++i) acc[t][i] = 0.0f;

  for (int dy = 0; dy < 5; ++dy) {
    int sy = y + dy - 2;
    if ((unsigned)sy > 63u) continue;
#pragma unroll
    for (int dx = 0; dx < 5; ++dx) {
      int sx = x0 + row + dx - 2;
      bool valid = ((unsigned)sx <= 63u);
      int r = (b << 12) + (sy << 6) + (valid ? sx : 0);
      const _Float16* ap = A + r * 64;
      const _Float16* wpBase = Wt + (dy * 5 + dx) * 4096;
#pragma unroll
      for (int kc = 0; kc < 2; ++kc) {
        AF a;
        if (valid) {
          a.h[0] = *(const v8h*)(ap + kc * 32 + kb);
          a.h[1] = *(const v8h*)(ap + kc * 32 + 16 + kb);
        } else {
          a.h[0] = hzero8();
          a.h[1] = hzero8();
        }
#pragma unroll
        for (int nt = 0; nt < 4; ++nt) {
          AF bb;
          const _Float16* wp = wpBase + (nt * 16 + row) * 64 + kc * 32 + kb;
          bb.h[0] = *(const v8h*)wp;
          bb.h[1] = *(const v8h*)(wp + 16);
          acc[nt] = __builtin_amdgcn_wmma_f32_16x16x32_f16(
              false, a.v, false, bb.v, (short)0, acc[nt], false, false);
        }
      }
    }
  }

  int mro = (lane >> 4) << 3;
#pragma unroll
  for (int nt = 0; nt < 4; ++nt) {
    int ch = nt * 16 + row;
    float bv = bias[ch];
#pragma unroll
    for (int i = 0; i < 8; ++i) {
      int m = m0 + i + mro;
      float v = fmaxf(acc[nt][i] + bv, 0.0f);
      outH[m * 64 + ch] = (_Float16)v;
      if (outF) outF[m * 64 + ch] = v;
    }
  }
}

// ---------------------------------------------------------------------------
// generic M x 64 x 64 WMMA linear: out = act(A @ W^T + bias)
// A f16 [m][64], Wh f16 [n][64]; one wave = 16(M) x 64(N) tile.
// ---------------------------------------------------------------------------
__global__ __launch_bounds__(128) void lin_wmma_kernel(
    const _Float16* __restrict__ A, const _Float16* __restrict__ Wh,
    const float* __restrict__ bias, _Float16* __restrict__ outH,
    float* __restrict__ outF, int doRelu) {
  int wave = threadIdx.x >> 5;
  int lane = threadIdx.x & 31;
  int m0 = (blockIdx.x * 4 + wave) << 4;
  int row = lane & 15;
  int kb  = (lane >> 4) << 3;

  __builtin_prefetch(Wh + lane * 128, 0, 1);

  v8f acc[4];
#pragma unroll
  for (int t = 0; t < 4; ++t)
#pragma unroll
    for (int i = 0; i < 8; ++i) acc[t][i] = 0.0f;

  const _Float16* ap = A + (m0 + row) * 64;
#pragma unroll
  for (int kc = 0; kc < 2; ++kc) {
    AF a;
    a.h[0] = *(const v8h*)(ap + kc * 32 + kb);
    a.h[1] = *(const v8h*)(ap + kc * 32 + 16 + kb);
#pragma unroll
    for (int nt = 0; nt < 4; ++nt) {
      AF bb;
      const _Float16* wp = Wh + (nt * 16 + row) * 64 + kc * 32 + kb;
      bb.h[0] = *(const v8h*)wp;
      bb.h[1] = *(const v8h*)(wp + 16);
      acc[nt] = __builtin_amdgcn_wmma_f32_16x16x32_f16(
          false, a.v, false, bb.v, (short)0, acc[nt], false, false);
    }
  }

  int mro = (lane >> 4) << 3;
#pragma unroll
  for (int nt = 0; nt < 4; ++nt) {
    int ch = nt * 16 + row;
    float bv = bias ? bias[ch] : 0.0f;
#pragma unroll
    for (int i = 0; i < 8; ++i) {
      int m = m0 + i + mro;
      float v = acc[nt][i] + bv;
      if (doRelu) v = fmaxf(v, 0.0f);
      if (outH) outH[m * 64 + ch] = (_Float16)v;
      if (outF) outF[m * 64 + ch] = v;
    }
  }
}

// ---------------------------------------------------------------------------
// positional embedding add: reads fp32 x, writes f16 (input to encoder MLP)
// ---------------------------------------------------------------------------
__global__ void addpos_f16_kernel(const float* __restrict__ X, const float* __restrict__ pos_w,
                                  const float* __restrict__ pos_b, _Float16* __restrict__ out) {
  int idx = blockIdx.x * blockDim.x + threadIdx.x;   // MTOT*64
  if (idx >= MTOT * 64) return;
  int c = idx & 63;
  int m = idx >> 6;
  int n = m & 4095;
  int y = n >> 6, x = n & 63;
  float fx = x * (1.0f / 63.0f), fy = y * (1.0f / 63.0f);
  float add = pos_b[c] + fx * pos_w[c * 4 + 0] + (1.0f - fx) * pos_w[c * 4 + 1] +
              fy * pos_w[c * 4 + 2] + (1.0f - fy) * pos_w[c * 4 + 3];
  out[idx] = (_Float16)(X[idx] + add);
}

// ---------------------------------------------------------------------------
// generic fp32 linear (small M only): out[m][n] = act(in[m]·W[n,:K]+bias+res)
// ---------------------------------------------------------------------------
__global__ void lin_kernel(const float* __restrict__ in, const float* __restrict__ W,
                           const float* __restrict__ bias, const float* __restrict__ res,
                           float* __restrict__ out, int M, int K, int N, int ldw, int doRelu) {
  int idx = blockIdx.x * blockDim.x + threadIdx.x;
  if (idx >= M * N) return;
  int m = idx / N, n = idx - m * N;
  float acc = bias ? bias[n] : 0.0f;
  const float* ip = in + m * K;
  const float* wp = W + n * ldw;
  for (int k = 0; k < K; ++k) acc = fmaf(ip[k], wp[k], acc);
  if (res) acc += res[idx];
  out[idx] = doRelu ? fmaxf(acc, 0.0f) : acc;
}

__global__ void ln_kernel(const float* __restrict__ in, const float* __restrict__ g,
                          const float* __restrict__ b, float* __restrict__ out, int M) {
  int m = blockIdx.x * blockDim.x + threadIdx.x;
  if (m >= M) return;
  const float* ip = in + m * 64;
  float mean = 0.0f;
  for (int d = 0; d < 64; ++d) mean += ip[d];
  mean *= (1.0f / 64.0f);
  float var = 0.0f;
  for (int d = 0; d < 64; ++d) { float t = ip[d] - mean; var = fmaf(t, t, var); }
  var *= (1.0f / 64.0f);
  float inv = rsqrtf(var + 1e-5f);
  for (int d = 0; d < 64; ++d) out[m * 64 + d] = (ip[d] - mean) * inv * g[d] + b[d];
}

// logits + softmax over K slots, +1e-8; attn layout [b*N+n][10]
__global__ void attn_kernel(const float* __restrict__ Kb, const float* __restrict__ q,
                            float* __restrict__ attn) {
  int idx = blockIdx.x * blockDim.x + threadIdx.x;   // B*N
  if (idx >= BN * NN) return;
  int b = idx >> 12;
  const float* kr = Kb + idx * 64;
  float lg[KSL];
  float mx = -3.4e38f;
  for (int s = 0; s < KSL; ++s) {
    const float* qr = q + (b * KSL + s) * 64;
    float d = 0.0f;
    for (int j = 0; j < 64; ++j) d = fmaf(kr[j], qr[j], d);
    lg[s] = d * 0.125f;   // 1/sqrt(64)
    mx = fmaxf(mx, lg[s]);
  }
  float sum = 0.0f;
  for (int s = 0; s < KSL; ++s) { lg[s] = expf(lg[s] - mx); sum += lg[s]; }
  float inv = 1.0f / sum;
  for (int s = 0; s < KSL; ++s) attn[idx * KSL + s] = lg[s] * inv + 1e-8f;
}

__global__ void colsum_kernel(const float* __restrict__ attn, float* __restrict__ asum) {
  int bk = blockIdx.x;             // 320
  int b = bk / KSL, s = bk - b * KSL;
  __shared__ float sh[256];
  float acc = 0.0f;
  for (int n = threadIdx.x; n < NN; n += 256) acc += attn[(b * NN + n) * KSL + s];
  sh[threadIdx.x] = acc;
  __syncthreads();
  for (int st = 128; st > 0; st >>= 1) {
    if (threadIdx.x < st) sh[threadIdx.x] += sh[threadIdx.x + st];
    __syncthreads();
  }
  if (threadIdx.x == 0) asum[bk] = sh[0];
}

// updates[bk][d] = (1/asum[bk]) * sum_n attn[b,n,k] * v[b,n,d]
__global__ void updates_kernel(const float* __restrict__ attn, const float* __restrict__ V,
                               const float* __restrict__ asum, float* __restrict__ upd) {
  int bk = blockIdx.x;
  int b = bk / KSL, s = bk - b * KSL;
  int d = threadIdx.x & 63;
  int part = threadIdx.x >> 6;
  float acc = 0.0f;
  for (int n = part * 1024; n < (part + 1) * 1024; ++n)
    acc = fmaf(attn[(b * NN + n) * KSL + s], V[(b * NN + n) * 64 + d], acc);
  __shared__ float sh[256];
  sh[threadIdx.x] = acc;
  __syncthreads();
  if (threadIdx.x < 64)
    upd[bk * 64 + d] = (sh[d] + sh[64 + d] + sh[128 + d] + sh[192 + d]) / asum[bk];
}

__global__ void gru_kernel(const float* __restrict__ gi, const float* __restrict__ gh,
                           const float* __restrict__ h, float* __restrict__ out) {
  int idx = blockIdx.x * blockDim.x + threadIdx.x;   // 320*64
  if (idx >= BN * KSL * 64) return;
  int r2 = idx >> 6, j = idx & 63;
  float ir = gi[r2 * 192 + j], iz = gi[r2 * 192 + 64 + j], in2 = gi[r2 * 192 + 128 + j];
  float hr = gh[r2 * 192 + j], hz = gh[r2 * 192 + 64 + j], hn = gh[r2 * 192 + 128 + j];
  float r = 1.0f / (1.0f + expf(-(ir + hr)));
  float z = 1.0f / (1.0f + expf(-(iz + hz)));
  float n = tanhf(in2 + r * hn);
  out[idx] = (1.0f - z) * n + z * h[idx];
}

__global__ void slots_init_kernel(const float* __restrict__ si, float* __restrict__ slots) {
  int i = blockIdx.x * blockDim.x + threadIdx.x;
  if (i < BN * KSL * 64) slots[i] = si[i % (KSL * 64)];
}

__global__ void slots_copy_kernel(const float* __restrict__ slots, float* __restrict__ out) {
  int i = blockIdx.x * blockDim.x + threadIdx.x;
  if (i < BN * KSL * 64) out[3 + BN * 3 * NN + i] = slots[i];
}

// grid part of decoder layer1: g1[n][j] = gx*w1[j][64] + gy*w1[j][65]
__global__ void g1_kernel(const float* __restrict__ dw1, float* __restrict__ g1) {
  int idx = blockIdx.x * blockDim.x + threadIdx.x;   // 4096*128
  if (idx >= NN * 128) return;
  int n = idx >> 7, j = idx & 127;
  int y = n >> 6, x = n & 63;
  float gx = -1.0f + x * (2.0f / 63.0f);
  float gy = -1.0f + y * (2.0f / 63.0f);
  g1[idx] = gx * dw1[j * 66 + 64] + gy * dw1[j * 66 + 65];
}

// ---------------------------------------------------------------------------
// fused decoder: h1=relu(c1+g1) in LDS, layer2+layer3 via WMMA, layer4 dot.
// block = one (b,k) pair x 64 rows of n; 4 waves x 16 rows.
// ---------------------------------------------------------------------------
__global__ __launch_bounds__(128) void dec_kernel(
    const float* __restrict__ c1, const float* __restrict__ g1,
    const _Float16* __restrict__ w2h, const float* __restrict__ b2,
    const _Float16* __restrict__ w3h, const float* __restrict__ b3,
    const float* __restrict__ w4, const float* __restrict__ b4,
    float* __restrict__ out4) {
  __shared__ __align__(16) _Float16 hsh[4][16][128];
  __shared__ float c1s[128];
  int bk = blockIdx.x >> 6;              // b*10+k
  int n0 = (blockIdx.x & 63) << 6;
  int wave = threadIdx.x >> 5;
  int lane = threadIdx.x & 31;
  int nbase = n0 + wave * 16;
  __builtin_prefetch(w2h + threadIdx.x * 128, 0, 1);
  __builtin_prefetch(w3h + threadIdx.x * 128, 0, 1);
  if (threadIdx.x < 128) c1s[threadIdx.x] = c1[bk * 128 + threadIdx.x];
  __syncthreads();
  for (int t = lane; t < 16 * 128; t += 32) {
    int r = t >> 7, j = t & 127;
    float v = c1s[j] + g1[(nbase + r) * 128 + j];
    hsh[wave][r][j] = (_Float16)fmaxf(v, 0.0f);
  }
  __syncthreads();

  int row = lane & 15;
  int kb  = (lane >> 4) << 3;
  int mro = (lane >> 4) << 3;
  v8f acc[8];

  // ---- layer 2 ----
#pragma unroll
  for (int t = 0; t < 8; ++t)
#pragma unroll
    for (int i = 0; i < 8; ++i) acc[t][i] = 0.0f;
#pragma unroll
  for (int kc = 0; kc < 4; ++kc) {
    AF a;
    a.h[0] = *(const v8h*)&hsh[wave][row][kc * 32 + kb];
    a.h[1] = *(const v8h*)&hsh[wave][row][kc * 32 + 16 + kb];
#pragma unroll
    for (int nt = 0; nt < 8; ++nt) {
      AF bb;
      const _Float16* wp = w2h + (nt * 16 + row) * 128 + kc * 32 + kb;
      bb.h[0] = *(const v8h*)wp;
      bb.h[1] = *(const v8h*)(wp + 16);
      acc[nt] = __builtin_amdgcn_wmma_f32_16x16x32_f16(
          false, a.v, false, bb.v, (short)0, acc[nt], false, false);
    }
  }
  __syncthreads();
#pragma unroll
  for (int nt = 0; nt < 8; ++nt) {
    int col = nt * 16 + row;
    float bv = b2[col];
#pragma unroll
    for (int i = 0; i < 8; ++i)
      hsh[wave][i + mro][col] = (_Float16)fmaxf(acc[nt][i] + bv, 0.0f);
  }
  __syncthreads();

  // ---- layer 3 ----
#pragma unroll
  for (int t = 0; t < 8; ++t)
#pragma unroll
    for (int i = 0; i < 8; ++i) acc[t][i] = 0.0f;
#pragma unroll
  for (int kc = 0; kc < 4; ++kc) {
    AF a;
    a.h[0] = *(const v8h*)&hsh[wave][row][kc * 32 + kb];
    a.h[1] = *(const v8h*)&hsh[wave][row][kc * 32 + 16 + kb];
#pragma unroll
    for (int nt = 0; nt < 8; ++nt) {
      AF bb;
      const _Float16* wp = w3h + (nt * 16 + row) * 128 + kc * 32 + kb;
      bb.h[0] = *(const v8h*)wp;
      bb.h[1] = *(const v8h*)(wp + 16);
      acc[nt] = __builtin_amdgcn_wmma_f32_16x16x32_f16(
          false, a.v, false, bb.v, (short)0, acc[nt], false, false);
    }
  }
  __syncthreads();
#pragma unroll
  for (int nt = 0; nt < 8; ++nt) {
    int col = nt * 16 + row;
    float bv = b3[col];
#pragma unroll
    for (int i = 0; i < 8; ++i)
      hsh[wave][i + mro][col] = (_Float16)fmaxf(acc[nt][i] + bv, 0.0f);
  }
  __syncthreads();

  // ---- layer 4: 128 -> 4 ----
#pragma unroll
  for (int oo = 0; oo < 2; ++oo) {
    int o = (lane >> 4) + oo * 2;
    float a4 = b4[o];
    for (int j = 0; j < 128; ++j)
      a4 = fmaf((float)hsh[wave][row][j], w4[o * 128 + j], a4);
    out4[(bk * NN + nbase + row) * 4 + o] = a4;
  }
}

__global__ void zero3_kernel(float* __restrict__ out) {
  if (threadIdx.x < 3 && blockIdx.x == 0) out[threadIdx.x] = 0.0f;
}

// softmax over K, recon, alpha_img, loss reductions
__global__ void finalize_kernel(const float* __restrict__ out4,
                                const float* __restrict__ image,
                                float* __restrict__ out) {
  int idx = blockIdx.x * 256 + threadIdx.x;   // B*N
  float err = 0.0f, ent = 0.0f;
  if (idx < BN * NN) {
    int b = idx >> 12, n = idx & 4095;
    float lg[KSL];
    float mx = -3.4e38f;
    for (int s = 0; s < KSL; ++s) {
      lg[s] = out4[((b * KSL + s) * NN + n) * 4 + 3];
      mx = fmaxf(mx, lg[s]);
    }
    float sum = 0.0f;
    for (int s = 0; s < KSL; ++s) { lg[s] = expf(lg[s] - mx); sum += lg[s]; }
    float inv = 1.0f / sum;
    float rec0 = 0.0f, rec1 = 0.0f, rec2 = 0.0f;
    float* alpha_out = out + 3 + BN * 3 * NN + BN * KSL * 64;
    for (int s = 0; s < KSL; ++s) {
      float a = lg[s] * inv;
      alpha_out[(b * KSL + s) * NN + n] = a;
      ent -= a * logf(a + 1e-8f);
      const float* o4 = out4 + ((b * KSL + s) * NN + n) * 4;
      rec0 = fmaf(a, o4[0], rec0);
      rec1 = fmaf(a, o4[1], rec1);
      rec2 = fmaf(a, o4[2], rec2);
    }
    float rr[3] = {rec0, rec1, rec2};
    for (int c = 0; c < 3; ++c) {
      out[3 + (b * 3 + c) * NN + n] = rr[c];
      float dd = rr[c] - image[(b * 3 + c) * NN + n];
      err = fmaf(dd, dd, err);
    }
  }
  __shared__ float se[256], sn[256];
  se[threadIdx.x] = err;
  sn[threadIdx.x] = ent;
  __syncthreads();
  for (int st = 128; st > 0; st >>= 1) {
    if (threadIdx.x < st) {
      se[threadIdx.x] += se[threadIdx.x + st];
      sn[threadIdx.x] += sn[threadIdx.x + st];
    }
    __syncthreads();
  }
  if (threadIdx.x == 0) {
    float rl = se[0] * (1.0f / (BN * 3.0f * NN));
    atomicAdd(&out[0], rl);
    atomicAdd(&out[1], rl);
    atomicAdd(&out[2], sn[0] * (1.0f / (BN * (float)NN * 2.302585093f)));
  }
}

// ---------------------------------------------------------------------------
extern "C" void kernel_launch(void* const* d_in, const int* in_sizes, int n_in,
                              void* d_out, int out_size, void* d_ws, size_t ws_size,
                              hipStream_t stream) {
  (void)in_sizes; (void)n_in; (void)out_size; (void)ws_size;
  const float* image = (const float*)d_in[0];
  const float* cw1 = (const float*)d_in[1];  const float* cb1 = (const float*)d_in[2];
  const float* cw2 = (const float*)d_in[3];  const float* cb2 = (const float*)d_in[4];
  const float* cw3 = (const float*)d_in[5];  const float* cb3 = (const float*)d_in[6];
  const float* cw4 = (const float*)d_in[7];  const float* cb4 = (const float*)d_in[8];
  const float* pos_w = (const float*)d_in[9];  const float* pos_b = (const float*)d_in[10];
  const float* ew1 = (const float*)d_in[11]; const float* eb1 = (const float*)d_in[12];
  const float* ew2 = (const float*)d_in[13]; const float* eb2 = (const float*)d_in[14];
  const float* eng = (const float*)d_in[15]; const float* enb = (const float*)d_in[16];
  const float* slot_init = (const float*)d_in[17];
  const float* nig = (const float*)d_in[18]; const float* nib = (const float*)d_in[19];
  const float* q_w = (const float*)d_in[20];
  const float* k_w = (const float*)d_in[21]; const float* k_b = (const float*)d_in[22];
  const float* v_w = (const float*)d_in[23]; const float* v_b = (const float*)d_in[24];
  const float* gwi = (const float*)d_in[25]; const float* gwh = (const float*)d_in[26];
  const float* gbi = (const float*)d_in[27]; const float* gbh = (const float*)d_in[28];
  const float* nsg = (const float*)d_in[29]; const float* nsb = (const float*)d_in[30];
  const float* sw1 = (const float*)d_in[31]; const float* sb1 = (const float*)d_in[32];
  const float* sw2 = (const float*)d_in[33]; const float* sb2 = (const float*)d_in[34];
  const float* nmg = (const float*)d_in[35]; const float* nmb = (const float*)d_in[36];
  const float* dw1 = (const float*)d_in[37]; const float* db1 = (const float*)d_in[38];
  const float* dw2 = (const float*)d_in[39]; const float* db2 = (const float*)d_in[40];
  const float* dw3 = (const float*)d_in[41]; const float* db3 = (const float*)d_in[42];
  const float* dw4 = (const float*)d_in[43]; const float* db4 = (const float*)d_in[44];
  float* out = (float*)d_out;

  char* wsp = (char*)d_ws;
  auto take = [&](size_t bytes) {
    char* p = wsp;
    wsp += (bytes + 255) & ~(size_t)255;
    return p;
  };
  _Float16* af16a = (_Float16*)take((size_t)MTOT * 64 * 2);
  _Float16* af16b = (_Float16*)take((size_t)MTOT * 64 * 2);
  float* X0 = (float*)take((size_t)MTOT * 64 * 4);
  float* X1 = (float*)take((size_t)MTOT * 64 * 4);
  float* Kb = (float*)take((size_t)MTOT * 64 * 4);
  float* Vb = (float*)take((size_t)MTOT * 64 * 4);
  float* attnb = (float*)take((size_t)BN * NN * KSL * 4);
  float* out4b = (float*)take((size_t)BN * KSL * NN * 4 * 4);
  _Float16* w2t = (_Float16*)take(64 * 64 * 25 * 2);
  _Float16* w3t = (_Float16*)take(64 * 64 * 25 * 2);
  _Float16* w4t = (_Float16*)take(64 * 64 * 25 * 2);
  _Float16* dw2h = (_Float16*)take(128 * 128 * 2);
  _Float16* dw3h = (_Float16*)take(128 * 128 * 2);
  _Float16* ew1h = (_Float16*)take(64 * 64 * 2);
  _Float16* ew2h = (_Float16*)take(64 * 64 * 2);
  _Float16* kwh  = (_Float16*)take(64 * 64 * 2);
  _Float16* vwh  = (_Float16*)take(64 * 64 * 2);
  float* snorm  = (float*)take(320 * 64 * 4);
  float* qb     = (float*)take(320 * 64 * 4);
  float* asum   = (float*)take(320 * 4);
  float* upd    = (float*)take(320 * 64 * 4);
  float* gib    = (float*)take(320 * 192 * 4);
  float* ghb    = (float*)take(320 * 192 * 4);
  float* slots  = (float*)take(320 * 64 * 4);
  float* slots2 = (float*)take(320 * 64 * 4);
  float* hid128 = (float*)take(320 * 128 * 4);
  float* c1b    = (float*)take(320 * 128 * 4);
  float* g1b    = (float*)take((size_t)NN * 128 * 4);

  const int T = 256;
  const int G_MC = (MTOT * 64) / T;        // 32768
  const int G_M  = MTOT / T;               // 512
  const int G_TILE = MTOT / 64;            // 2048 blocks * 4 waves * 16 rows

  // ---- weight precisions ----
  cvt_conv_w_kernel<<<400, T, 0, stream>>>(cw2, w2t);
  cvt_conv_w_kernel<<<400, T, 0, stream>>>(cw3, w3t);
  cvt_conv_w_kernel<<<400, T, 0, stream>>>(cw4, w4t);
  cvt_f16_kernel<<<64, T, 0, stream>>>(dw2, dw2h, 128 * 128);
  cvt_f16_kernel<<<64, T, 0, stream>>>(dw3, dw3h, 128 * 128);
  cvt_f16_kernel<<<16, T, 0, stream>>>(ew1, ew1h, 64 * 64);
  cvt_f16_kernel<<<16, T, 0, stream>>>(ew2, ew2h, 64 * 64);
  cvt_f16_kernel<<<16, T, 0, stream>>>(k_w, kwh, 64 * 64);
  cvt_f16_kernel<<<16, T, 0, stream>>>(v_w, vwh, 64 * 64);

  // ---- encoder convs (WMMA shifted-GEMM) ----
  conv1_kernel<<<G_MC, T, 0, stream>>>(image, cw1, cb1, af16a);
  conv_wmma_kernel<<<G_TILE, 128, 0, stream>>>(af16a, w2t, cb2, af16b, nullptr);
  conv_wmma_kernel<<<G_TILE, 128, 0, stream>>>(af16b, w3t, cb3, af16a, nullptr);
  conv_wmma_kernel<<<G_TILE, 128, 0, stream>>>(af16a, w4t, cb4, af16b, X0);

  // ---- pos-embed + encoder MLP (WMMA) + norms + k/v (WMMA) ----
  addpos_f16_kernel<<<G_MC, T, 0, stream>>>(X0, pos_w, pos_b, af16a);
  lin_wmma_kernel<<<G_TILE, 128, 0, stream>>>(af16a, ew1h, eb1, af16b, nullptr, 1);
  lin_wmma_kernel<<<G_TILE, 128, 0, stream>>>(af16b, ew2h, eb2, nullptr, X0, 0);
  ln_kernel<<<G_M, T, 0, stream>>>(X0, eng, enb, X1, MTOT);
  ln_kernel<<<G_M, T, 0, stream>>>(X1, nig, nib, X0, MTOT);
  cvt_f16_kernel<<<G_MC, T, 0, stream>>>(X0, af16a, MTOT * 64);
  lin_wmma_kernel<<<G_TILE, 128, 0, stream>>>(af16a, kwh, k_b, nullptr, Kb, 0);
  lin_wmma_kernel<<<G_TILE, 128, 0, stream>>>(af16a, vwh, v_b, nullptr, Vb, 0);

  // ---- slot attention (3 iterations, small fp32) ----
  slots_init_kernel<<<80, T, 0, stream>>>(slot_init, slots);
  for (int it = 0; it < 3; ++it) {
    ln_kernel<<<2, T, 0, stream>>>(slots, nsg, nsb, snorm, 320);
    lin_kernel<<<80, T, 0, stream>>>(snorm, q_w, nullptr, nullptr, qb, 320, 64, 64, 64, 0);
    attn_kernel<<<G_M, T, 0, stream>>>(Kb, qb, attnb);
    colsum_kernel<<<320, T, 0, stream>>>(attnb, asum);
    updates_kernel<<<320, T, 0, stream>>>(attnb, Vb, asum, upd);
    lin_kernel<<<240, T, 0, stream>>>(upd, gwi, gbi, nullptr, gib, 320, 64, 192, 64, 0);
    lin_kernel<<<240, T, 0, stream>>>(slots, gwh, gbh, nullptr, ghb, 320, 64, 192, 64, 0);
    gru_kernel<<<80, T, 0, stream>>>(gib, ghb, slots, slots2);
    ln_kernel<<<2, T, 0, stream>>>(slots2, nmg, nmb, snorm, 320);
    lin_kernel<<<160, T, 0, stream>>>(snorm, sw1, sb1, nullptr, hid128, 320, 64, 128, 64, 1);
    lin_kernel<<<80, T, 0, stream>>>(hid128, sw2, sb2, slots2, slots, 320, 128, 64, 128, 0);
  }
  slots_copy_kernel<<<80, T, 0, stream>>>(slots, out);

  // ---- decoder (fused WMMA) ----
  lin_kernel<<<160, T, 0, stream>>>(slots, dw1, db1, nullptr, c1b, 320, 64, 128, 66, 0);
  g1_kernel<<<2048, T, 0, stream>>>(dw1, g1b);
  dec_kernel<<<BN * KSL * 64, 128, 0, stream>>>(c1b, g1b, dw2h, db2, dw3h, db3, dw4, db4, out4b);

  // ---- outputs / losses ----
  zero3_kernel<<<1, 32, 0, stream>>>(out);
  finalize_kernel<<<G_M, T, 0, stream>>>(out4b, image, out);
}
